// TGNN_9809705304189
// MI455X (gfx1250) — compile-verified, matrix operand order checked by reading, not run on hardware
//
#include <hip/hip_runtime.h>
#include <math.h>

// ---------------------------------------------------------------------------
// TGNN for MI455X (gfx1250, wave32).
//
// Dominant op per (b,t) graph:  h_nodes = softmax(masked scores) @ H.
// Since scores[i][j] = src_i + dst_j + b, the src_i + b terms cancel in the
// row softmax, so  alpha[i][j] = mask_ij * e_j / sum_j mask_ij * e_j  with
// e_j = exp(dst_j).  Hence  h_nodes = (M @ G) / (M @ e)  with binary M and
// G[j][c] = e_j * H[j][c].  M is exact in bf16 (0/1), and G is carried as a
// bf16 hi+lo split (two v_wmma_f32_16x16x32_bf16 per tile into one f32
// accumulator) for near-f32 accuracy at bf16-WMMA throughput.
// dst_j = H_j . a2 is O(1) for this input distribution (|dst| << 80), so
// exp() needs no max-shift.
//
// CDNA5 paths used: v_wmma_f32_16x16x32_bf16, global_load_async_to_lds_b32
// (+ s_wait_asynccnt), global_prefetch_b8, ds atomics, split wait counters.
// ---------------------------------------------------------------------------

typedef __attribute__((ext_vector_type(16))) __bf16 v16bf;
typedef __attribute__((ext_vector_type(8)))  float  v8f;

#define N_NODES  1024
#define HD       32
#define T_STEPS  32
#define B_BATCH  4
#define THRESH_F 150.0f
#define PASSES   4
#define WIN      (N_NODES / PASSES)   // 256 j per pass
#define CHUNKS   (WIN / 32)           // 8 K-chunks of 32 per pass
#define NTHREADS 512                  // 16 wave32

__global__ __launch_bounds__(NTHREADS)
void tgnn_gat_kernel(const float* __restrict__ X,      // [128][1024][4]
                     const float* __restrict__ Wf,     // [4][32]
                     const float* __restrict__ bfv,    // [32]
                     const float* __restrict__ a_attn, // [64] (use second half)
                     float* __restrict__ pooled)       // [128][32]
{
    // B-fragment layout: frag[chunk][ct][lane][e], lane = half*16 + N,
    // element e of a lane maps to K = (e>=8)*16 + half*8 + (e&7).
    __shared__ __align__(32) __bf16 frag_hi[CHUNKS][2][32][16]; // 16 KB
    __shared__ __align__(32) __bf16 frag_lo[CHUNKS][2][32][16]; // 16 KB
    __shared__ float d_sh[N_NODES];                             // 4 KB
    __shared__ float e_sh[N_NODES];                             // 4 KB
    __shared__ float pool_sh[HD];

    const int g    = blockIdx.x;          // graph = b*T + t (matches X layout)
    const int tid  = threadIdx.x;
    const int lane = tid & 31;
    const int wave = tid >> 5;
    const int hlf  = lane >> 4;           // which 16-lane half
    const int Mrow = lane & 15;
    const float* __restrict__ Xg = X + (size_t)g * N_NODES * 4;
    const float* __restrict__ a2 = a_attn + HD;

    // ---- phase 0: distances (= last feature) for all nodes, pool init ----
    // Strided gather X[n][3] -> LDS via the gfx1250 async-to-LDS DMA path:
    // per-lane global address + per-lane LDS address, tracked by ASYNCcnt.
#pragma unroll
    for (int r = 0; r < N_NODES / NTHREADS; ++r) {
        const int n = tid + r * NTHREADS;
        const unsigned lds_addr = (unsigned)(uintptr_t)(void*)&d_sh[n];
        const float* gp = &Xg[n * 4 + 3];
        asm volatile("global_load_async_to_lds_b32 %0, %1, off"
                     :: "v"(lds_addr), "v"(gp) : "memory");
    }
    if (tid < HD) pool_sh[tid] = 0.0f;
    asm volatile("s_wait_asynccnt 0" ::: "memory");
    __syncthreads();

    // Persistent per-wave state: 4 row-tiles (64 tiles / 16 waves).
    v8f   acc[4][2];
    float den[4];
    float di[4];
#pragma unroll
    for (int q = 0; q < 4; ++q) {
        acc[q][0] = {}; acc[q][1] = {}; den[q] = 0.0f;
        di[q] = d_sh[(wave * 4 + q) * 16 + Mrow];
    }

#pragma unroll 1
    for (int pass = 0; pass < PASSES; ++pass) {
        const int nb = pass * WIN;

        // ---- producer: H, dst, e, G fragments for j in [nb, nb+WIN) ----
        if (tid < WIN) {
            const int n = nb + tid;
            const float x0 = Xg[n*4+0], x1 = Xg[n*4+1],
                        x2 = Xg[n*4+2], x3 = Xg[n*4+3];
            float h[HD];
            float dstv = 0.0f;
#pragma unroll
            for (int c = 0; c < HD; ++c) {
                float hv = bfv[c] + x0*Wf[0*HD+c] + x1*Wf[1*HD+c]
                                  + x2*Wf[2*HD+c] + x3*Wf[3*HD+c];
                h[c] = hv;
                dstv += hv * a2[c];
            }
            const float ev = __expf(dstv);   // bounded; max-shift cancels anyway
            e_sh[n] = ev;

            const int kin  = tid & 31;                       // j within chunk
            const int cw   = tid >> 5;                       // chunk 0..7
            const int eIdx = ((kin & 16) ? 8 : 0) + (kin & 7);
            const int hB   = (kin >> 3) & 1;
#pragma unroll
            for (int c = 0; c < HD; ++c) {
                const float gval = ev * h[c];
                const __bf16 ghi = (__bf16)gval;
                const __bf16 glo = (__bf16)(gval - (float)ghi);
                const int ct = c >> 4, Nc = c & 15;
                frag_hi[cw][ct][hB * 16 + Nc][eIdx] = ghi;
                frag_lo[cw][ct][hB * 16 + Nc][eIdx] = glo;
            }
        } else if (pass + 1 < PASSES && tid < 2 * WIN) {
            // prefetch next pass's X window (global_prefetch_b8)
            __builtin_prefetch(&Xg[(nb + WIN + (tid - WIN)) * 4], 0, 0);
        }
        __syncthreads();

        // ---- consumer: masked GEMM tiles over this j-window ----
#pragma unroll 1
        for (int cw = 0; cw < CHUNKS; ++cw) {
            const int jj = nb + cw * 32;
            const v16bf bhi0 = *(const v16bf*)&frag_hi[cw][0][lane][0];
            const v16bf blo0 = *(const v16bf*)&frag_lo[cw][0][lane][0];
            const v16bf bhi1 = *(const v16bf*)&frag_hi[cw][1][lane][0];
            const v16bf blo1 = *(const v16bf*)&frag_lo[cw][1][lane][0];

            float dj[16], ej[16];
#pragma unroll
            for (int e2 = 0; e2 < 16; ++e2) {
                const int k = ((e2 >= 8) ? 16 : 0) + hlf * 8 + (e2 & 7);
                dj[e2] = d_sh[jj + k];
                ej[e2] = e_sh[jj + k];
            }
#pragma unroll
            for (int q = 0; q < 4; ++q) {
                v16bf a;
                float dsum = 0.0f;
#pragma unroll
                for (int e2 = 0; e2 < 16; ++e2) {
                    // self-loop (i==j) is implied: |d_i - d_i| = 0 < THRESH
                    const bool m = fabsf(di[q] - dj[e2]) < THRESH_F;
                    a[e2] = (__bf16)(m ? 1.0f : 0.0f);
                    dsum += m ? ej[e2] : 0.0f;
                }
                den[q] += dsum;
                acc[q][0] = __builtin_amdgcn_wmma_f32_16x16x32_bf16(
                    false, a, false, bhi0, (short)0, acc[q][0], false, false);
                acc[q][0] = __builtin_amdgcn_wmma_f32_16x16x32_bf16(
                    false, a, false, blo0, (short)0, acc[q][0], false, false);
                acc[q][1] = __builtin_amdgcn_wmma_f32_16x16x32_bf16(
                    false, a, false, bhi1, (short)0, acc[q][1], false, false);
                acc[q][1] = __builtin_amdgcn_wmma_f32_16x16x32_bf16(
                    false, a, false, blo1, (short)0, acc[q][1], false, false);
            }
        }
        __syncthreads();  // fragments may be overwritten next pass
    }

    // ---- normalize rows by denom and pool (mean over nodes) ----
    float pool0 = 0.0f, pool1 = 0.0f;
#pragma unroll
    for (int q = 0; q < 4; ++q) {
        // lanes L and L^16 hold complementary K-halves of row (L&15)
        const float dfull = den[q] + __shfl_xor(den[q], 16, 32);
#pragma unroll
        for (int c = 0; c < 8; ++c) {
            // D element c of this lane is row (c + 8*hlf) of the tile
            const float dr = __shfl(dfull, c + hlf * 8, 32);
            pool0 += acc[q][0][c] / dr;
            pool1 += acc[q][1][c] / dr;
        }
    }
    atomicAdd(&pool_sh[Mrow],      pool0);   // column N = lane&15
    atomicAdd(&pool_sh[16 + Mrow], pool1);   // column 16 + N
    __syncthreads();
    if (tid < HD) pooled[g * HD + tid] = pool_sh[tid] * (1.0f / N_NODES);
}

// ---------------------------------------------------------------------------
// GRU over T=32 steps + output head. Tiny: one block, thread = (b, k).
// ---------------------------------------------------------------------------
__global__ __launch_bounds__(128)
void tgnn_gru_kernel(const float* __restrict__ pooled, // [B*T][32]
                     const float* __restrict__ W_ih,   // [96][32]
                     const float* __restrict__ W_hh,   // [96][32]
                     const float* __restrict__ b_ih,   // [96]
                     const float* __restrict__ b_hh,   // [96]
                     const float* __restrict__ Wo,     // [32]
                     const float* __restrict__ bo,     // [1]
                     float* __restrict__ out)          // [B]
{
    __shared__ float h_sh[B_BATCH][HD];
    __shared__ float x_sh[B_BATCH][HD];
    const int tid = threadIdx.x;
    const int b = tid >> 5, k = tid & 31;
    h_sh[b][k] = 0.0f;
    __syncthreads();

    for (int t = 0; t < T_STEPS; ++t) {
        x_sh[b][k] = pooled[(b * T_STEPS + t) * HD + k];
        __syncthreads();
        float ir = b_ih[k], iz = b_ih[32 + k], inn = b_ih[64 + k];
        float hr = b_hh[k], hz = b_hh[32 + k], hnn = b_hh[64 + k];
#pragma unroll
        for (int c = 0; c < HD; ++c) {
            const float xv = x_sh[b][c], hv = h_sh[b][c];
            ir  += xv * W_ih[k * HD + c];
            iz  += xv * W_ih[(32 + k) * HD + c];
            inn += xv * W_ih[(64 + k) * HD + c];
            hr  += hv * W_hh[k * HD + c];
            hz  += hv * W_hh[(32 + k) * HD + c];
            hnn += hv * W_hh[(64 + k) * HD + c];
        }
        const float r = 1.0f / (1.0f + __expf(-(ir + hr)));
        const float z = 1.0f / (1.0f + __expf(-(iz + hz)));
        const float n = tanhf(inn + r * hnn);
        const float hnew = (1.0f - z) * n + z * h_sh[b][k];
        __syncthreads();
        h_sh[b][k] = hnew;
        __syncthreads();
    }
    if (k == 0) {
        float o = bo[0];
        for (int c = 0; c < HD; ++c) o += h_sh[b][c] * Wo[c];
        out[b] = o;
    }
}

extern "C" void kernel_launch(void* const* d_in, const int* in_sizes, int n_in,
                              void* d_out, int out_size, void* d_ws, size_t ws_size,
                              hipStream_t stream) {
    (void)in_sizes; (void)n_in; (void)out_size; (void)ws_size;
    const float* X      = (const float*)d_in[0];
    const float* Wf     = (const float*)d_in[1];
    const float* bfv    = (const float*)d_in[2];
    const float* a_attn = (const float*)d_in[3];
    // d_in[4] = b_attn: cancels inside the row softmax, unused.
    const float* W_ih   = (const float*)d_in[5];
    const float* W_hh   = (const float*)d_in[6];
    const float* b_ih   = (const float*)d_in[7];
    const float* b_hh   = (const float*)d_in[8];
    const float* Wo     = (const float*)d_in[9];
    const float* bo     = (const float*)d_in[10];

    float* pooled = (float*)d_ws;                 // [128][32] f32 = 16 KB
    float* out    = (float*)d_out;                // [4]

    tgnn_gat_kernel<<<B_BATCH * T_STEPS, NTHREADS, 0, stream>>>(
        X, Wf, bfv, a_attn, pooled);
    tgnn_gru_kernel<<<1, 128, 0, stream>>>(
        pooled, W_ih, W_hh, b_ih, b_hh, Wo, bo, out);
}
